// SelfAttentionCartesianRPE_32856499815191
// MI455X (gfx1250) — compile-verified, matrix operand order checked by reading, not run on hardware
//
#include <hip/hip_runtime.h>
#include <hip/hip_bf16.h>

// ---------------------------------------------------------------------------
// Self-attention with Cartesian RPE bias — CDNA5 (gfx1250), wave32 + WMMA.
// f16 data paths everywhere a WMMA consumes data; f32 accumulate.
// ---------------------------------------------------------------------------

typedef __attribute__((ext_vector_type(16))) _Float16 v16h;
typedef __attribute__((ext_vector_type(8)))  float    v8f;

#define BATCH   2
#define SEQ     512
#define DIM     512
#define HEADS   8
#define DHEAD   64
#define INNER   (HEADS * DHEAD)      // 512
#define QKVCOLS (3 * INNER)          // 1536

// ---------------------------------------------------------------------------
// WMMA helpers: 16x16x32 f16 -> f32.
// A-matrix per-lane layout (ISA 7.12.2, 16-bit A 16x32):
//   lane L holds row m = L & 15; half element e maps to
//   K(e,L) = (e < 8 ? e : e + 8) + (L >= 16 ? 8 : 0)   (two contiguous runs of 8)
// B-matrix (32x16): lane L holds column n = L & 15, same K(e,L) mapping.
// C/D: lane L holds column n = L & 15, VGPR e holds row m = e + (L>=16 ? 8:0).
// ---------------------------------------------------------------------------

__device__ __forceinline__ v8f wmma_step(v16h a, v16h b, v8f c) {
    return __builtin_amdgcn_wmma_f32_16x16x32_f16(
        /*neg_a=*/false, a, /*neg_b=*/false, b,
        /*c_mod=*/(short)0, c, /*reuse_a=*/false, /*reuse_b=*/false);
}

// Generic f16 row-major tile fetch (works for global or LDS pointers).
// Used for: A tiles (row = m), B^T tiles (row = n), pre-transposed weight
// tiles (row = n), and LDS-staged V^T tiles (row = n).  Each lane touches two
// contiguous 8-half (16B) runs -> b128 loads.
__device__ __forceinline__ v16h load_rm_h16(const _Float16* P, int ld,
                                            int row0, int k0, int lane) {
    v16h a;
    const _Float16* base = P + (size_t)(row0 + (lane & 15)) * ld + k0;
    int khi = (lane >= 16) ? 8 : 0;
#pragma unroll
    for (int e = 0; e < 16; ++e) {
        int k = (e < 8 ? e : e + 8) + khi;
        a[e] = base[k];
    }
    return a;
}

// A tile from row-major f32 (converted on load; compiler emits b128+cvt_pk).
__device__ __forceinline__ v16h load_rm_f32(const float* P, int ld,
                                            int row0, int k0, int lane) {
    v16h a;
    const float* base = P + (size_t)(row0 + (lane & 15)) * ld + k0;
    int khi = (lane >= 16) ? 8 : 0;
#pragma unroll
    for (int e = 0; e < 16; ++e) {
        int k = (e < 8 ? e : e + 8) + khi;
        a[e] = (_Float16)base[k];
    }
    return a;
}

// ---------------------------------------------------------------------------
// Kernel 0: pack f32 weights [K][N] -> f16 transposed [N][K].
// ---------------------------------------------------------------------------
__global__ __launch_bounds__(256)
void pack_w_kernel(const float* __restrict__ W, _Float16* __restrict__ Wt,
                   int K, int N) {
    int idx = blockIdx.x * 256 + threadIdx.x;
    if (idx >= K * N) return;
    int k = idx / N, n = idx - k * N;
    Wt[(size_t)n * K + k] = (_Float16)W[idx];
}

// ---------------------------------------------------------------------------
// Kernels 1 & 5: GEMM  C[M][N] = A[M][K] @ Bt[N][K]^T (+bias).
// A: f32 or f16 row-major; Bt: f16 pre-transposed; C: f16 or f32.
// One wave32 per 16x16 output tile.
// ---------------------------------------------------------------------------
template <bool AHALF, bool CHALF>
__global__ __launch_bounds__(256)
void gemm_bt_wmma(const void* __restrict__ Aptr,
                  const _Float16* __restrict__ Bt,
                  const float* __restrict__ bias,
                  void* __restrict__ Cptr,
                  int M, int Ncols, int K) {
    int wave = threadIdx.x >> 5;
    int lane = threadIdx.x & 31;
    int ntile = Ncols >> 4;
    int gw = blockIdx.x * 8 + wave;
    if (gw >= (M >> 4) * ntile) return;   // wave-uniform
    int tm = gw / ntile;
    int tn = gw - tm * ntile;
    int row0 = tm << 4, col0 = tn << 4;

    v8f acc = {};
    for (int kk = 0; kk < K; kk += 32) {
        if (kk + 32 < K) {
            if (AHALF)
                __builtin_prefetch((const _Float16*)Aptr + (size_t)row0 * K + kk + 32, 0, 1);
            else
                __builtin_prefetch((const float*)Aptr + (size_t)row0 * K + kk + 32, 0, 1);
            __builtin_prefetch(Bt + (size_t)col0 * K + kk + 32, 0, 1);
        }
        v16h a;
        if (AHALF) a = load_rm_h16((const _Float16*)Aptr, K, row0, kk, lane);
        else       a = load_rm_f32((const float*)Aptr, K, row0, kk, lane);
        v16h b = load_rm_h16(Bt, K, col0, kk, lane);   // B[k][n] = Bt[n][k]
        acc = wmma_step(a, b, acc);
    }

    int n = lane & 15;
    int mb = (lane >= 16) ? 8 : 0;
    float badd = bias ? bias[col0 + n] : 0.0f;
#pragma unroll
    for (int e = 0; e < 8; ++e) {
        size_t off = (size_t)(row0 + mb + e) * Ncols + col0 + n;
        if (CHALF) ((_Float16*)Cptr)[off] = (_Float16)(acc[e] + badd);
        else       ((float*)Cptr)[off]    = acc[e] + badd;
    }
}

// ---------------------------------------------------------------------------
// Kernel 2: RPE bias.  One thread per (b, i, j).  Fourier features computed
// on the fly (freqs = 1..32), MLP 130->16 (exact GELU) -> 8, weights in LDS.
// Writes bias[b][h][i][j].  FLOP hotspot; inherently VALU (K=130, sincos).
// ---------------------------------------------------------------------------
__global__ __launch_bounds__(256)
void rpe_bias_kernel(const float* __restrict__ pos,
                     const float* __restrict__ W1, const float* __restrict__ b1,
                     const float* __restrict__ W2, const float* __restrict__ b2,
                     float* __restrict__ bias) {
    __shared__ float sW1[130 * 16];
    __shared__ float sb1[16];
    __shared__ float sW2[16 * 8];
    __shared__ float sb2[8];
    for (int t = threadIdx.x; t < 130 * 16; t += 256) sW1[t] = W1[t];
    if (threadIdx.x < 16)  sb1[threadIdx.x] = b1[threadIdx.x];
    if (threadIdx.x < 128) sW2[threadIdx.x] = W2[threadIdx.x];
    if (threadIdx.x < 8)   sb2[threadIdx.x] = b2[threadIdx.x];
    __syncthreads();

    int t   = blockIdx.x * 256 + threadIdx.x;          // [0, B*N*N)
    int b   = t >> 18;                                 // / (512*512)
    int rem = t & (SEQ * SEQ - 1);
    int i   = rem >> 9;
    int j   = rem & (SEQ - 1);

    const float* pi = pos + ((size_t)b * SEQ + i) * 2;
    const float* pj = pos + ((size_t)b * SEQ + j) * 2;
    float dx = (pj[0] - pi[0]) * 0.1f;  dx = dx / (1.0f + fabsf(dx));
    float dy = (pj[1] - pi[1]) * 0.1f;  dy = dy / (1.0f + fabsf(dy));

    float h[16];
#pragma unroll
    for (int c = 0; c < 16; ++c)
        h[c] = sb1[c] + dx * sW1[0 * 16 + c] + dy * sW1[65 * 16 + c];

    const float PI = 3.14159265358979f;
    for (int kb = 0; kb < 32; ++kb) {
        float f = (float)(kb + 1) * PI;
        float sx, cx, sy, cy;
        __sincosf(dx * f, &sx, &cx);
        __sincosf(dy * f, &sy, &cy);
        const float* rsx = &sW1[(1 + kb) * 16];
        const float* rcx = &sW1[(33 + kb) * 16];
        const float* rsy = &sW1[(66 + kb) * 16];
        const float* rcy = &sW1[(98 + kb) * 16];
#pragma unroll
        for (int c = 0; c < 16; ++c)
            h[c] += sx * rsx[c] + cx * rcx[c] + sy * rsy[c] + cy * rcy[c];
    }

    float g[16];
#pragma unroll
    for (int c = 0; c < 16; ++c)
        g[c] = 0.5f * h[c] * (1.0f + erff(h[c] * 0.70710678118654752f));

#pragma unroll
    for (int hh = 0; hh < 8; ++hh) {
        float acc = sb2[hh];
#pragma unroll
        for (int c = 0; c < 16; ++c) acc += g[c] * sW2[c * 8 + hh];
        bias[((((size_t)b * HEADS + hh) * SEQ + i) << 9) + j] = acc;
    }
}

// ---------------------------------------------------------------------------
// Kernel 3: attention.  One block (8 waves) per (b, h, 16 query rows).
//  - scores strip 16x512 via WMMA (Q, K^T both f16, contiguous) into LDS,
//    fused *scale + bias; concurrently stage V panel (512x64) into LDS
//    transposed so the P@V B-tiles are contiguous LDS reads.
//  - wave-parallel softmax over 512-wide rows (wave32 shuffles), P -> f16 LDS.
//  - out = P @ V via WMMA (4 waves, one 16x16 head-dim tile each), f16 store.
// ---------------------------------------------------------------------------
#define VSTR 528   // padded row stride (halves) for V^T panel

__global__ __launch_bounds__(256)
void attention_kernel(const _Float16* __restrict__ qkv,
                      const float* __restrict__ bias,
                      _Float16* __restrict__ attn_out) {
    __shared__ float    s_sc[16][SEQ];    // 32 KB
    __shared__ _Float16 s_p [16][SEQ];    // 16 KB
    __shared__ _Float16 s_vt[DHEAD][VSTR]; // ~67.6 KB   (total ~117 KB / WGP)

    int wave = threadIdx.x >> 5;
    int lane = threadIdx.x & 31;

    int idx = blockIdx.x;                 // [0, B*H*32)
    int b   = idx / (HEADS * 32);
    int rm  = idx - b * (HEADS * 32);
    int h   = rm / 32;
    int i0  = (rm - h * 32) << 4;

    const _Float16* base   = qkv + (size_t)b * SEQ * QKVCOLS;
    const float*    biasBH = bias + (((size_t)b * HEADS + h) * SEQ) * SEQ;
    const float     scale  = 0.125f;      // 64^-0.5

    // ---- Stage V panel (512 x 64 f16) into LDS, transposed ----
    {
        const _Float16* Vb = base + 2 * INNER + h * DHEAD;
        for (int el = threadIdx.x; el < SEQ * DHEAD; el += 256) {
            int j = el >> 6, d = el & 63;
            s_vt[d][j] = Vb[(size_t)j * QKVCOLS + d];
        }
    }

    // ---- Phase 1: scores ----
    for (int tt = 0; tt < 4; ++tt) {
        int j0 = ((tt << 3) + wave) << 4;
        v8f acc = {};
#pragma unroll
        for (int kk = 0; kk < DHEAD; kk += 32) {
            v16h a  = load_rm_h16(base,         QKVCOLS, i0, h * DHEAD + kk, lane);
            v16h bb = load_rm_h16(base + INNER, QKVCOLS, j0, h * DHEAD + kk, lane);
            acc = wmma_step(a, bb, acc);
        }
        int n = lane & 15;
        int mb = (lane >= 16) ? 8 : 0;
#pragma unroll
        for (int e = 0; e < 8; ++e) {
            int m = mb + e;
            s_sc[m][j0 + n] = acc[e] * scale + biasBH[(size_t)(i0 + m) * SEQ + j0 + n];
        }
    }
    __syncthreads();

    // ---- Phase 2: softmax (each wave owns rows wave, wave+8) ----
    for (int r = wave; r < 16; r += 8) {
        float mx = -3.4e38f;
        for (int c = lane; c < SEQ; c += 32) mx = fmaxf(mx, s_sc[r][c]);
#pragma unroll
        for (int off = 16; off > 0; off >>= 1)
            mx = fmaxf(mx, __shfl_xor(mx, off, 32));
        float sum = 0.0f;
        for (int c = lane; c < SEQ; c += 32) {
            float e = __expf(s_sc[r][c] - mx);
            s_sc[r][c] = e;
            sum += e;
        }
#pragma unroll
        for (int off = 16; off > 0; off >>= 1)
            sum += __shfl_xor(sum, off, 32);
        float inv = 1.0f / sum;
        for (int c = lane; c < SEQ; c += 32)
            s_p[r][c] = (_Float16)(s_sc[r][c] * inv);
    }
    __syncthreads();

    // ---- Phase 3: out = P @ V (waves 0..3, 16 head-dims each) ----
    if (wave < 4) {
        int n0 = wave << 4;
        v8f acc = {};
        for (int kk = 0; kk < SEQ; kk += 32) {
            v16h a  = load_rm_h16(&s_p[0][0],  SEQ,  0,  kk, lane);
            v16h bb = load_rm_h16(&s_vt[0][0], VSTR, n0, kk, lane); // B[k][n]=Vt[n][k]
            acc = wmma_step(a, bb, acc);
        }
        int n = lane & 15;
        int mb = (lane >= 16) ? 8 : 0;
#pragma unroll
        for (int e = 0; e < 8; ++e) {
            int m = mb + e;
            attn_out[(size_t)(b * SEQ + i0 + m) * INNER + h * DHEAD + n0 + n] =
                (_Float16)acc[e];
        }
    }
}

// ---------------------------------------------------------------------------
// Launch
// ---------------------------------------------------------------------------
extern "C" void kernel_launch(void* const* d_in, const int* in_sizes, int n_in,
                              void* d_out, int out_size, void* d_ws, size_t ws_size,
                              hipStream_t stream) {
    const float* x         = (const float*)d_in[0];
    const float* positions = (const float*)d_in[1];
    const float* W_qkv     = (const float*)d_in[2];
    const float* W_out     = (const float*)d_in[3];
    const float* b_out     = (const float*)d_in[4];
    const float* rpe_W1    = (const float*)d_in[5];
    const float* rpe_b1    = (const float*)d_in[6];
    const float* rpe_W2    = (const float*)d_in[7];
    const float* rpe_b2    = (const float*)d_in[8];
    float* out = (float*)d_out;

    // Workspace carve-up (all chunks 16B aligned)
    char* w = (char*)d_ws;
    _Float16* ws_Wqkvt = (_Float16*)w; w += (size_t)QKVCOLS * DIM * 2;        // 1.5 MB
    _Float16* ws_Woutt = (_Float16*)w; w += (size_t)DIM * INNER * 2;          // 0.5 MB
    _Float16* ws_qkvh  = (_Float16*)w; w += (size_t)BATCH * SEQ * QKVCOLS * 2;// 3 MB
    float*    ws_bias  = (float*)w;    w += (size_t)BATCH * HEADS * SEQ * SEQ * 4; // 16 MB
    _Float16* ws_atth  = (_Float16*)w;                                        // 1 MB

    // 0) Pack weights -> f16 transposed
    pack_w_kernel<<<(DIM * QKVCOLS + 255) / 256, 256, 0, stream>>>(
        W_qkv, ws_Wqkvt, DIM, QKVCOLS);
    pack_w_kernel<<<(INNER * DIM + 255) / 256, 256, 0, stream>>>(
        W_out, ws_Woutt, INNER, DIM);

    // 1) QKV projection: [B*N, 512] @ [512, 1536] -> f16
    {
        int tiles = (BATCH * SEQ / 16) * (QKVCOLS / 16);   // 6144
        gemm_bt_wmma<false, true><<<tiles / 8, 256, 0, stream>>>(
            x, ws_Wqkvt, nullptr, ws_qkvh, BATCH * SEQ, QKVCOLS, DIM);
    }

    // 2) RPE bias: one thread per (b, i, j)
    rpe_bias_kernel<<<(BATCH * SEQ * SEQ) / 256, 256, 0, stream>>>(
        positions, rpe_W1, rpe_b1, rpe_W2, rpe_b2, ws_bias);

    // 3) Attention: one block per (b, h, 16 query rows)
    attention_kernel<<<BATCH * HEADS * (SEQ / 16), 256, 0, stream>>>(
        ws_qkvh, ws_bias, ws_atth);

    // 4) Output projection: [B*N, 512] @ [512, 512] + b_out -> f32
    {
        int tiles = (BATCH * SEQ / 16) * (DIM / 16);       // 2048
        gemm_bt_wmma<true, false><<<tiles / 8, 256, 0, stream>>>(
            ws_atth, ws_Woutt, b_out, out, BATCH * SEQ, DIM, INNER);
    }
}
